// RAgent_26663156973581
// MI455X (gfx1250) — matrix operand order repaired and checked by reading, not compiled, not run on hardware
//
#include <hip/hip_runtime.h>
#include <hip/hip_bf16.h>

typedef __attribute__((ext_vector_type(16))) _Float16 v16h;
typedef __attribute__((ext_vector_type(8)))  float    v8f;

#define B_ROWS   1048576
#define IN_DIM   25
#define HID      64
#define OUT_DIM  6
#define ROWS_PER_BLOCK 64
#define NWAVES   4                 // 128 threads per block

#define XSTRIDE  25                // contiguous tile; odd stride -> conflict-free gathers
#define HSTRIDE  68                // 272B rows = 17 x 16B: b128-aligned, 4-way on gathers
#define SSTRIDE  65                // transpose stage: gcd(65,64)=1, conflict-free

// Weight-fragment workspace layout: 32x16 f16 tiles, 512 halves (1KB) each.
#define T_PRE(j)      (j)                          // 4 tiles  (25->64, K padded to 32)
#define T_IH(g,j,ks)  (4  + (((g)*4+(j))*2+(ks)))  // 24 tiles (64->192, gates r,z,n)
#define T_HH(g,j,ks)  (28 + (((g)*4+(j))*2+(ks)))  // 24 tiles
#define N_TILES 52                                  // 52 KB total in d_ws

// gfx1250 async global->LDS DMA (ASYNCcnt-tracked), with safe fallback
#if defined(__gfx1250__) && __has_builtin(__builtin_amdgcn_global_load_async_to_lds_b128)
#define USE_ASYNC_COPY 1
#else
#define USE_ASYNC_COPY 0
#endif

// ---------------------------------------------------------------------------
// Prep: swizzle fp32 weights into per-lane f16 B-fragments (B is 32x16 tile:
// lane -> column n = lane&15, half = lane>>4; element e -> K = 16*half + e).
// ---------------------------------------------------------------------------
__global__ void prep_wfrag(const float* __restrict__ pre_w,
                           const float* __restrict__ w_ih,
                           const float* __restrict__ w_hh,
                           _Float16* __restrict__ wfrag)
{
    const int tile = blockIdx.x;
    const int lane = threadIdx.x;          // 32 threads
    const int n    = lane & 15;
    const int hb   = lane >> 4;
    _Float16* dst  = wfrag + (tile * 32 + lane) * 16;

    if (tile < 4) {                        // pre_w.T tiles (zero-pad K>=25)
        const int col = tile * 16 + n;
        #pragma unroll
        for (int e = 0; e < 16; ++e) {
            const int K = 16 * hb + e;
            const float v = (K < IN_DIM) ? pre_w[col * IN_DIM + K] : 0.0f;
            dst[e] = (_Float16)v;
        }
    } else if (tile < 28) {                // w_ih.T tiles
        const int t = tile - 4;
        const int ks = t & 1, j = (t >> 1) & 3, g = t >> 3;
        const int col = g * 64 + j * 16 + n;
        #pragma unroll
        for (int e = 0; e < 16; ++e) {
            const int K = 32 * ks + 16 * hb + e;
            dst[e] = (_Float16)w_ih[col * HID + K];
        }
    } else {                               // w_hh.T tiles
        const int t = tile - 28;
        const int ks = t & 1, j = (t >> 1) & 3, g = t >> 3;
        const int col = g * 64 + j * 16 + n;
        #pragma unroll
        for (int e = 0; e < 16; ++e) {
            const int K = 32 * ks + 16 * hb + e;
            dst[e] = (_Float16)w_hh[col * HID + K];
        }
    }
}

// A-fragment K index for 16-bit 16x32 A (lane half selects 16-wide contiguous K range)
__device__ __forceinline__ int kmapA(int e, int hl) {
    return 16 * hl + e;
}

__device__ __forceinline__ float sigm(float v) {
    return 1.0f / (1.0f + __expf(-v));
}

// ---------------------------------------------------------------------------
// Fused pre-Linear + GRU step + head + softmax. 16 rows per wave via WMMA.
// ---------------------------------------------------------------------------
__global__ void __launch_bounds__(128)
gru_fused(const float* __restrict__ x,
          const float* __restrict__ pre_b,
          const float* __restrict__ b_ih,
          const float* __restrict__ b_hh,
          const float* __restrict__ out_w,
          const float* __restrict__ out_b,
          const float* __restrict__ h0,
          const _Float16* __restrict__ wfrag,
          float* __restrict__ out)
{
    __shared__ __align__(16) float lds_x [ROWS_PER_BLOCK * XSTRIDE];
    __shared__ __align__(16) float lds_h0[ROWS_PER_BLOCK * HSTRIDE];
    __shared__ float lds_st[NWAVES][16][SSTRIDE];

    const int tid  = threadIdx.x;
    const int wave = tid >> 5;
    const int lane = tid & 31;
    const int ln   = lane & 15;
    const int hl   = lane >> 4;
    const long base = (long)blockIdx.x * ROWS_PER_BLOCK;

#if USE_ASYNC_COPY
    {   // DMA global->LDS without VGPR round trip (ASYNCcnt-tracked).
        // Builtin signature (from clang diagnostic): (int4 AS1*, int4 AS3*, Ii, Ii)
        typedef int v4i __attribute__((vector_size(16)));
        typedef __attribute__((address_space(1))) v4i gv4i;
        typedef __attribute__((address_space(3))) v4i lv4i;

        gv4i* gx = (gv4i*)(void*)const_cast<float*>(x + base * IN_DIM); // 6400B, 16B aligned
        lv4i* lx = (lv4i*)(void*)lds_x;
        for (int i = tid; i < (ROWS_PER_BLOCK * IN_DIM) / 4; i += 128)  // 400 x b128
            __builtin_amdgcn_global_load_async_to_lds_b128(gx + i, lx + i, 0, 0);

        gv4i* gh = (gv4i*)(void*)const_cast<float*>(h0 + base * HID);
        lv4i* lh = (lv4i*)(void*)lds_h0;
        for (int i = tid; i < ROWS_PER_BLOCK * (HID / 4); i += 128) {   // 1024 x b128
            const int r = i >> 4, c16 = i & 15;
            __builtin_amdgcn_global_load_async_to_lds_b128(gh + (r * 16 + c16),
                                                           lh + (r * 17 + c16), 0, 0);
        }
#if __has_builtin(__builtin_amdgcn_s_wait_asynccnt)
        __builtin_amdgcn_s_wait_asynccnt(0);
#else
        asm volatile("s_wait_asynccnt 0x0" ::: "memory");
#endif
    }
#else
    for (int i = tid; i < ROWS_PER_BLOCK * IN_DIM; i += 128)
        lds_x[i] = x[base * IN_DIM + i];                        // tile is contiguous
    for (int i = tid; i < ROWS_PER_BLOCK * HID; i += 128) {
        const int r = i >> 6, c = i & 63;
        lds_h0[r * HSTRIDE + c] = h0[(base + r) * HID + c];
    }
#endif
    __syncthreads();

    const int wr = wave * 16;              // wave's row offset within block

    // ---- Phase A: xp = x @ pre_w.T + pre_b  (16x64), staged in LDS ----
    v16h a_x;
    #pragma unroll
    for (int e = 0; e < 16; ++e) {
        const int K = kmapA(e, hl);
        a_x[e] = (K < IN_DIM) ? (_Float16)lds_x[(wr + ln) * XSTRIDE + K] : (_Float16)0.0f;
    }
    #pragma unroll
    for (int j = 0; j < 4; ++j) {
        const v16h bf = *reinterpret_cast<const v16h*>(wfrag + (T_PRE(j) * 32 + lane) * 16);
        const float bb = pre_b[16 * j + ln];
        v8f c;
        #pragma unroll
        for (int v = 0; v < 8; ++v) c[v] = bb;
        v8f acc = __builtin_amdgcn_wmma_f32_16x16x32_f16(false, a_x, false, bf,
                                                         (short)0, c, false, false);
        #pragma unroll
        for (int v = 0; v < 8; ++v)
            lds_st[wave][v + 8 * hl][16 * j + ln] = acc[v];   // C layout: M=v+8*half, N=ln
    }

    // ---- Phase B: A-fragments of xp and h0 (16x64 -> two K=32 fragments each) ----
    v16h a_xp[2], a_h0[2];
    #pragma unroll
    for (int ks = 0; ks < 2; ++ks) {
        #pragma unroll
        for (int e = 0; e < 16; ++e) {
            const int K = 32 * ks + kmapA(e, hl);
            a_xp[ks][e] = (_Float16)lds_st[wave][ln][K];
            a_h0[ks][e] = (_Float16)lds_h0[(wr + ln) * HSTRIDE + K];
        }
    }

    // ---- Phase C: gates (r,z,n) and h, one 16-column tile at a time ----
    #pragma unroll
    for (int j = 0; j < 4; ++j) {
        const int nG = 16 * j + ln;
        v8f gi[3], gh[3];
        #pragma unroll
        for (int g = 0; g < 3; ++g) {
            {   // gi_g = xp @ w_ih.T tile + b_ih
                const v16h b0 = *reinterpret_cast<const v16h*>(wfrag + (T_IH(g, j, 0) * 32 + lane) * 16);
                const v16h b1 = *reinterpret_cast<const v16h*>(wfrag + (T_IH(g, j, 1) * 32 + lane) * 16);
                const float bb = b_ih[g * 64 + nG];
                v8f c;
                #pragma unroll
                for (int v = 0; v < 8; ++v) c[v] = bb;
                c     = __builtin_amdgcn_wmma_f32_16x16x32_f16(false, a_xp[0], false, b0, (short)0, c, false, false);
                gi[g] = __builtin_amdgcn_wmma_f32_16x16x32_f16(false, a_xp[1], false, b1, (short)0, c, false, false);
            }
            {   // gh_g = h0 @ w_hh.T tile + b_hh
                const v16h b0 = *reinterpret_cast<const v16h*>(wfrag + (T_HH(g, j, 0) * 32 + lane) * 16);
                const v16h b1 = *reinterpret_cast<const v16h*>(wfrag + (T_HH(g, j, 1) * 32 + lane) * 16);
                const float bb = b_hh[g * 64 + nG];
                v8f c;
                #pragma unroll
                for (int v = 0; v < 8; ++v) c[v] = bb;
                c     = __builtin_amdgcn_wmma_f32_16x16x32_f16(false, a_h0[0], false, b0, (short)0, c, false, false);
                gh[g] = __builtin_amdgcn_wmma_f32_16x16x32_f16(false, a_h0[1], false, b1, (short)0, c, false, false);
            }
        }
        // Elementwise gate math directly on C-layout accumulators
        #pragma unroll
        for (int v = 0; v < 8; ++v) {
            const float r  = sigm(gi[0][v] + gh[0][v]);
            const float z  = sigm(gi[1][v] + gh[1][v]);
            float pn = gi[2][v] + r * gh[2][v];
            pn = fminf(fmaxf(pn, -15.0f), 15.0f);
            const float t  = __expf(2.0f * pn);
            const float nn = (t - 1.0f) / (t + 1.0f);            // tanh
            const float hp = lds_h0[(wr + v + 8 * hl) * HSTRIDE + nG];
            lds_st[wave][v + 8 * hl][nG] = (1.0f - z) * nn + z * hp;
        }
    }

    // ---- Phase D: head (64->6) + softmax, one row per lane ----
    if (lane < 16) {
        const long grow = base + wr + lane;
        float logit[OUT_DIM];
        #pragma unroll
        for (int o = 0; o < OUT_DIM; ++o) {
            float s = out_b[o];
            for (int k = 0; k < HID; ++k)
                s += lds_st[wave][lane][k] * out_w[o * HID + k]; // out_w reads are wave-uniform
            logit[o] = s;
        }
        float mx = logit[0];
        #pragma unroll
        for (int o = 1; o < OUT_DIM; ++o) mx = fmaxf(mx, logit[o]);
        float es[OUT_DIM], sum = 0.0f;
        #pragma unroll
        for (int o = 0; o < OUT_DIM; ++o) { es[o] = __expf(logit[o] - mx); sum += es[o]; }
        const float inv = 1.0f / sum;
        #pragma unroll
        for (int o = 0; o < OUT_DIM; ++o)
            out[grow * OUT_DIM + o] = es[o] * inv;
    }
}

// ---------------------------------------------------------------------------
extern "C" void kernel_launch(void* const* d_in, const int* in_sizes, int n_in,
                              void* d_out, int out_size, void* d_ws, size_t ws_size,
                              hipStream_t stream)
{
    const float* x     = (const float*)d_in[0];
    const float* pre_w = (const float*)d_in[1];
    const float* pre_b = (const float*)d_in[2];
    const float* w_ih  = (const float*)d_in[3];
    const float* w_hh  = (const float*)d_in[4];
    const float* b_ih  = (const float*)d_in[5];
    const float* b_hh  = (const float*)d_in[6];
    const float* out_w = (const float*)d_in[7];
    const float* out_b = (const float*)d_in[8];
    const float* h0    = (const float*)d_in[9];
    float* out = (float*)d_out;

    _Float16* wfrag = (_Float16*)d_ws;     // needs 52 KB of workspace

    prep_wfrag<<<N_TILES, 32, 0, stream>>>(pre_w, w_ih, w_hh, wfrag);

    const int nblocks = B_ROWS / ROWS_PER_BLOCK;   // 16384
    gru_fused<<<nblocks, 128, 0, stream>>>(x, pre_b, b_ih, b_hh,
                                           out_w, out_b, h0, wfrag, out);
}
